// MixtureCDFCoupling_12532714570305
// MI455X (gfx1250) — matrix operand order repaired and checked by reading, not compiled, not run on hardware
//
#include <hip/hip_runtime.h>

#define B_      16
#define L_      1024
#define C_      64
#define K_      10
#define H_      1024
#define COUT_   2048          // C * (2 + 3K) = 64 * 32
#define MTILE   32            // rows per workgroup (2 WMMA row-blocks)
#define NTHREADS 256
#define NWAVES   8
#define CHUNKW  512           // N columns per GEMM2 chunk
#define NCHUNK  (COUT_ / CHUNKW)     // 4
#define NT      4             // N-tiles per wave per chunk (wave covers 64 cols)
#define CH_PER_CHUNK (CHUNKW / 32)   // 16 channels per chunk
#define PLSTRIDE (CH_PER_CHUNK * 33) // padded row stride in floats (528)

// packed weight sizes (f16, fragment-ordered: [kblk][n][16])
#define W2P_ELEMS  (COUT_ * (H_ / 16) * 16)   // 2048 cols * 64 kblks * 16
#define W1P_ELEMS  (H_ * (C_ / 16) * 16)      // 1024 cols * 4 kblks * 16
#define W2P_BYTES  (W2P_ELEMS * 2)
#define W1P_BYTES  (W1P_ELEMS * 2)

typedef __attribute__((ext_vector_type(16))) _Float16 v16h;
typedef __attribute__((ext_vector_type(8)))  _Float16 v8h;
typedef __attribute__((ext_vector_type(8)))  float    v8f;

__global__ void init_ldj_kernel(float* __restrict__ ldj) {
    if (threadIdx.x < B_) ldj[threadIdx.x] = 0.0f;
}

// w2 (H x COUT f32, row-major) -> w2p[kblk][n][i] = (f16) w2[(kblk*16+i)*COUT + n]
__global__ __launch_bounds__(NTHREADS)
void repack_w2_kernel(const float* __restrict__ w2, _Float16* __restrict__ w2p) {
    const int idx  = blockIdx.x * NTHREADS + threadIdx.x;   // 0 .. 64*2048-1
    const int kblk = idx >> 11;                             // / COUT_
    const int n    = idx & (COUT_ - 1);
    union { v8h v[2]; _Float16 h[16]; } u;
    #pragma unroll
    for (int i = 0; i < 16; ++i)
        u.h[i] = (_Float16)w2[(kblk * 16 + i) * COUT_ + n];
    v8h* dst = (v8h*)&w2p[idx << 4];
    dst[0] = u.v[0];
    dst[1] = u.v[1];
}

// w1 (C x H f32, row-major) -> w1p[kblk][n][i] = (f16) w1[(kblk*16+i)*H + n]
__global__ __launch_bounds__(NTHREADS)
void repack_w1_kernel(const float* __restrict__ w1, _Float16* __restrict__ w1p) {
    const int idx  = blockIdx.x * NTHREADS + threadIdx.x;   // 0 .. 4*1024-1
    const int kblk = idx >> 10;                             // / H_
    const int n    = idx & (H_ - 1);
    union { v8h v[2]; _Float16 h[16]; } u;
    #pragma unroll
    for (int i = 0; i < 16; ++i)
        u.h[i] = (_Float16)w1[(kblk * 16 + i) * H_ + n];
    v8h* dst = (v8h*)&w1p[idx << 4];
    dst[0] = u.v[0];
    dst[1] = u.v[1];
}

__device__ __forceinline__ float softplus_f(float x) {
    return fmaxf(x, 0.0f) + log1pf(expf(-fabsf(x)));
}
__device__ __forceinline__ float safe_log_f(float x) {
    return logf(fmaxf(x, 1e-22f));
}
__device__ __forceinline__ float gelu_f(float x) {
    const float k0 = 0.7978845608028654f;  // sqrt(2/pi)
    return 0.5f * x * (1.0f + tanhf(k0 * (x + 0.044715f * x * x * x)));
}

template <bool PACKED>
__global__ __launch_bounds__(NTHREADS)
void fused_mixture_cdf_kernel(const float* __restrict__ z,   const float* __restrict__ mask,
                              const float* __restrict__ w1,  const float* __restrict__ b1,
                              const float* __restrict__ w2,  const float* __restrict__ b2,
                              const float* __restrict__ sfv, const float* __restrict__ msfv,
                              const _Float16* __restrict__ w1p, const _Float16* __restrict__ w2p,
                              float* __restrict__ zout, float* __restrict__ ldj)
{
    __shared__ _Float16 Zh[MTILE * C_];        // masked z tile (A of GEMM1)     4 KB
    __shared__ float    Zf[MTILE * C_];        // raw z tile                     8 KB
    __shared__ _Float16 Hl[MTILE * H_];        // gelu activations, f16         64 KB
    __shared__ float    Pl[MTILE * PLSTRIDE];  // nn_out chunk, padded f32      66 KB
    __shared__ float    s_ldj;

    const int tid  = threadIdx.x;
    const int lane = tid & 31;
    const int wave = tid >> 5;
    const int half = lane >> 4;
    const int l15  = lane & 15;
    const int r0   = blockIdx.x * MTILE;     // global row base (row = b*L + l)
    const int batch = r0 / L_;

    if (tid == 0) s_ldj = 0.0f;

    // ---- load z tile: raw f32 + masked f16 ----
    for (int i = tid; i < MTILE * C_; i += NTHREADS) {
        int row = i / C_, c = i % C_;
        float v = z[(r0 + row) * C_ + c];
        Zf[i] = v;
        Zh[i] = (_Float16)(v * mask[c]);
    }
    __syncthreads();

    // ---- GEMM1: Hl = gelu(Zin @ W1 + b1); 2 passes of 512 cols ----
    for (int npass = 0; npass < 2; ++npass) {
        v8f acc[2][NT];
        #pragma unroll
        for (int rb = 0; rb < 2; ++rb)
            #pragma unroll
            for (int t = 0; t < NT; ++t) acc[rb][t] = (v8f){};

        #pragma unroll
        for (int ks = 0; ks < 2; ++ks) {
            union { v16h v; v8h p[2]; _Float16 h[16]; } a[2];
            #pragma unroll
            for (int rb = 0; rb < 2; ++rb) {
                const int kb = ks * 32 + half * 16;
                const _Float16* src = &Zh[(rb * 16 + l15) * C_ + kb];
                a[rb].p[0] = *(const v8h*)src;
                a[rb].p[1] = *(const v8h*)(src + 8);
            }
            const int kblk = ks * 2 + half;                 // 0..3
            #pragma unroll
            for (int t = 0; t < NT; ++t) {
                const int ncol = npass * 512 + wave * 64 + t * 16 + l15;
                union { v16h v; v8h p[2]; _Float16 h[16]; } bf;
                if (PACKED) {
                    const _Float16* src = &w1p[(size_t)((kblk << 10) + ncol) << 4];
                    bf.p[0] = *(const v8h*)src;
                    bf.p[1] = *(const v8h*)(src + 8);
                } else {
                    const int kb = kblk * 16;
                    #pragma unroll
                    for (int v2 = 0; v2 < 8; ++v2) {
                        bf.h[2 * v2]     = (_Float16)w1[(kb + 2 * v2) * H_ + ncol];
                        bf.h[2 * v2 + 1] = (_Float16)w1[(kb + 2 * v2 + 1) * H_ + ncol];
                    }
                }
                #pragma unroll
                for (int rb = 0; rb < 2; ++rb)
                    acc[rb][t] = __builtin_amdgcn_wmma_f32_16x16x32_f16(
                        false, a[rb].v, false, bf.v, (short)0, acc[rb][t], false, false);
            }
        }
        #pragma unroll
        for (int t = 0; t < NT; ++t) {
            const int ncol = npass * 512 + wave * 64 + t * 16 + l15;
            const float bias = b1[ncol];
            #pragma unroll
            for (int rb = 0; rb < 2; ++rb) {
                #pragma unroll
                for (int r = 0; r < 8; ++r) {
                    const int m = rb * 16 + r + 8 * half;
                    Hl[m * H_ + ncol] = (_Float16)gelu_f(acc[rb][t][r] + bias);
                }
            }
        }
    }
    __syncthreads();

    float ldj_acc = 0.0f;

    for (int chunk = 0; chunk < NCHUNK; ++chunk) {
        // ---- GEMM2: Pl = Hl @ W2[:, chunk*512 : +512] + b2 ----
        {
            v8f acc[2][NT];
            #pragma unroll
            for (int rb = 0; rb < 2; ++rb)
                #pragma unroll
                for (int t = 0; t < NT; ++t) acc[rb][t] = (v8f){};

            for (int ks = 0; ks < 32; ++ks) {
                union { v16h v; v8h p[2]; _Float16 h[16]; } a[2];
                #pragma unroll
                for (int rb = 0; rb < 2; ++rb) {
                    const int kb = ks * 32 + half * 16;
                    const _Float16* src = &Hl[(rb * 16 + l15) * H_ + kb];
                    a[rb].p[0] = *(const v8h*)src;
                    a[rb].p[1] = *(const v8h*)(src + 8);
                }
                const int kblk = ks * 2 + half;             // 0..63
                #pragma unroll
                for (int t = 0; t < NT; ++t) {
                    const int col_g = chunk * CHUNKW + wave * 64 + t * 16 + l15;  // 0..2047
                    union { v16h v; v8h p[2]; _Float16 h[16]; } bf;
                    if (PACKED) {
                        const _Float16* src = &w2p[(size_t)((kblk << 11) + col_g) << 4];
                        bf.p[0] = *(const v8h*)src;
                        bf.p[1] = *(const v8h*)(src + 8);
                    } else {
                        const int kb = kblk * 16;
                        #pragma unroll
                        for (int v2 = 0; v2 < 8; ++v2) {
                            bf.h[2 * v2]     = (_Float16)w2[(kb + 2 * v2) * COUT_ + col_g];
                            bf.h[2 * v2 + 1] = (_Float16)w2[(kb + 2 * v2 + 1) * COUT_ + col_g];
                        }
                    }
                    #pragma unroll
                    for (int rb = 0; rb < 2; ++rb)
                        acc[rb][t] = __builtin_amdgcn_wmma_f32_16x16x32_f16(
                            false, a[rb].v, false, bf.v, (short)0, acc[rb][t], false, false);
                }
            }
            #pragma unroll
            for (int t = 0; t < NT; ++t) {
                const int col_l = wave * 64 + t * 16 + l15;        // 0..511
                const int ch    = col_l >> 5;                      // 0..15
                const int jj    = col_l & 31;
                const float bias = b2[chunk * CHUNKW + col_l];
                #pragma unroll
                for (int rb = 0; rb < 2; ++rb) {
                    #pragma unroll
                    for (int r = 0; r < 8; ++r) {
                        const int m = rb * 16 + r + 8 * half;
                        Pl[m * PLSTRIDE + ch * 33 + jj] = acc[rb][t][r] + bias;
                    }
                }
            }
        }
        __syncthreads();

        // ---- elementwise mixture-CDF math for channels [chunk*16, +16) ----
        for (int p = tid; p < MTILE * CH_PER_CHUNK; p += NTHREADS) {
            const int row = p >> 4;
            const int cl  = p & (CH_PER_CHUNK - 1);
            const int c   = chunk * CH_PER_CHUNK + cl;
            const float* par = &Pl[row * PLSTRIDE + cl * 33];  // 32 params of (row, c)
            const float zval = Zf[row * C_ + c];
            const float mk   = mask[c];
            const float cm   = 1.0f - mk;

            const float tpar  = par[0] * cm;
            const float sf    = expf(sfv[c]);
            const float log_s = tanhf(par[1] / fmaxf(sf, 1.0f)) * sf * cm;

            float lp[K_], uu[K_], mls[K_];
            float maxlp = -INFINITY;
            #pragma unroll
            for (int k = 0; k < K_; ++k) {
                const float lpi = par[2 + k] * cm;
                const float mt  = par[2 + K_ + k] * cm;
                const float msf = expf(msfv[c * K_ + k]);
                const float ml  = tanhf(par[2 + 2 * K_ + k] / fmaxf(msf, 1.0f)) * msf * cm;
                lp[k]  = lpi;
                mls[k] = ml;
                uu[k]  = (zval - mt) * expf(-ml);
                maxlp  = fmaxf(maxlp, lpi);
            }
            float sum = 0.0f;
            #pragma unroll
            for (int k = 0; k < K_; ++k) sum += expf(lp[k] - maxlp);
            const float lse_pi = maxlp + logf(sum);

            float a1[K_], a2[K_];
            float m1 = -INFINITY, m2 = -INFINITY;
            #pragma unroll
            for (int k = 0; k < K_; ++k) {
                const float lw = lp[k] - lse_pi;           // log_softmax
                const float ls = -softplus_f(-uu[k]);      // log_sigmoid(u)
                a1[k] = lw + ls;
                a2[k] = lw + uu[k] - mls[k] - 2.0f * softplus_f(uu[k]);
                m1 = fmaxf(m1, a1[k]);
                m2 = fmaxf(m2, a2[k]);
            }
            float s1 = 0.0f, s2 = 0.0f;
            #pragma unroll
            for (int k = 0; k < K_; ++k) { s1 += expf(a1[k] - m1); s2 += expf(a2[k] - m2); }
            const float log_cdf      = m1 + logf(s1);
            const float logistic_ldj = m2 + logf(s2);

            const float p_cdf    = expf(log_cdf);
            const float z_logit  = -safe_log_f(1.0f / p_cdf - 1.0f);
            const float mixt_ldj = -safe_log_f(p_cdf) - safe_log_f(1.0f - p_cdf);

            float zo = (z_logit + tpar) * expf(log_s);
            zo = zo * cm + zval * mk;
            zout[(r0 + row) * C_ + c] = zo;
            ldj_acc += cm * (log_s + mixt_ldj + logistic_ldj);
        }
        __syncthreads();
    }

    atomicAdd(&s_ldj, ldj_acc);
    __syncthreads();
    if (tid == 0) atomicAdd(&ldj[batch], s_ldj);
}

extern "C" void kernel_launch(void* const* d_in, const int* in_sizes, int n_in,
                              void* d_out, int out_size, void* d_ws, size_t ws_size,
                              hipStream_t stream) {
    (void)in_sizes; (void)n_in; (void)out_size;
    const float* z    = (const float*)d_in[0];
    const float* mask = (const float*)d_in[1];
    const float* w1   = (const float*)d_in[2];
    const float* b1   = (const float*)d_in[3];
    const float* w2   = (const float*)d_in[4];
    const float* b2   = (const float*)d_in[5];
    const float* sfv  = (const float*)d_in[6];
    const float* msfv = (const float*)d_in[7];

    float* zout = (float*)d_out;
    float* ldj  = zout + (size_t)B_ * L_ * C_;

    init_ldj_kernel<<<1, 64, 0, stream>>>(ldj);

    const size_t needed = (size_t)W2P_BYTES + (size_t)W1P_BYTES;
    const bool packed = (d_ws != nullptr) && (ws_size >= needed);

    if (packed) {
        _Float16* w2p = (_Float16*)d_ws;
        _Float16* w1p = (_Float16*)((char*)d_ws + W2P_BYTES);
        repack_w2_kernel<<<(64 * COUT_) / NTHREADS, NTHREADS, 0, stream>>>(w2, w2p);
        repack_w1_kernel<<<(4 * H_) / NTHREADS, NTHREADS, 0, stream>>>(w1, w1p);
        fused_mixture_cdf_kernel<true><<<(B_ * L_) / MTILE, NTHREADS, 0, stream>>>(
            z, mask, w1, b1, w2, b2, sfv, msfv, w1p, w2p, zout, ldj);
    } else {
        fused_mixture_cdf_kernel<false><<<(B_ * L_) / MTILE, NTHREADS, 0, stream>>>(
            z, mask, w1, b1, w2, b2, sfv, msfv, nullptr, nullptr, zout, ldj);
    }
}
